// TransformerCouplingLayer_23545010716750
// MI455X (gfx1250) — matrix operand order repaired and checked by reading, not compiled, not run on hardware
//
#include <hip/hip_runtime.h>

// ---------------- problem constants (match reference) ----------------
static constexpr int LAYERS = 6;
static constexpr int C      = 512;
static constexpr int HALF   = 256;
static constexpr int FF     = 2048;
static constexpr int KS     = 3;
static constexpr int H      = 8;
static constexpr int DK     = C / H;   // 64
static constexpr int WIN    = 4;
static constexpr int BATCH  = 4;
static constexpr int T      = 1024;
static constexpr float EPS  = 1e-5f;

// ---------------- WMMA types ----------------
typedef __attribute__((ext_vector_type(16))) __bf16          v16bf;
typedef __attribute__((ext_vector_type(8)))  float           v8f;
typedef __attribute__((ext_vector_type(8)))  unsigned short  u16x8;
typedef __attribute__((ext_vector_type(4)))  float           f32x4;

union FragBF { v16bf v; u16x8 h[2]; };

__device__ __forceinline__ unsigned short f2bf(float f) {
  union { __bf16 b; unsigned short u; } x;
  x.b = (__bf16)f;                      // native v_cvt on gfx1250
  return x.u;
}

__device__ __forceinline__ u16x8 pack8(f32x4 a, f32x4 b) {
  u16x8 p;
  p[0] = f2bf(a.x); p[1] = f2bf(a.y); p[2] = f2bf(a.z); p[3] = f2bf(a.w);
  p[4] = f2bf(b.x); p[5] = f2bf(b.y); p[6] = f2bf(b.z); p[7] = f2bf(b.w);
  return p;
}

// =====================================================================
// bf16 WMMA GEMM, REQUIRES: M % 128 == 0, N % 64 == 0, K % 32 == 0.
//   C[z,m,n] = alpha * sum_k A[z,m,k]*B[z,k,n] (+bias[m]) (+relu)
// aMode: 0 = A contiguous along k (sAk==1), 1 = A contiguous along m (sAm==1)
// bMode: 0 = B contiguous along n (sBn==1), 1 = B contiguous along k (sBk==1),
//        2 = conv3 im2col (k = c*3+j, col = n+j-1, zero-pad outside [0,convN))
// Block tile 128(M) x 64(N); 8 waves = 4(M-groups of 32) x 2(N-groups of 32);
// each wave: 2 A-frags x 2 B-frags -> 4 WMMAs per 32-wide K-step.
// =====================================================================
__global__ __launch_bounds__(256)
void gemm_bf16_wmma(const float* __restrict__ A, long sAm, long sAk, long sAz,
                    const float* __restrict__ B, long sBk, long sBn, long sBz,
                    const float* __restrict__ bias,
                    float* __restrict__ Cmat, long sCm, long sCn, long sCz,
                    int M, int N, int K,
                    float alpha, int relu, int aMode, int bMode, int convN)
{
  __shared__ unsigned short Alds[128 * 40];  // [m][k], row stride 40 ushorts (80B)
  __shared__ unsigned short Blds[64 * 40];   // [n][k] (transposed)

  const int tid  = threadIdx.x;
  const int lane = tid & 31;
  const int wave = tid >> 5;
  const int mGrp = wave & 3;       // 4 M groups of 32 (2 x 16)
  const int nGrp = wave >> 2;      // 2 N groups of 32 (2 x 16)
  const int half = lane >> 4;
  const int l15  = lane & 15;

  const int  mBase = blockIdx.y * 128;
  const int  nBase = blockIdx.x * 64;
  const long az = (long)blockIdx.z * sAz;
  const long bz = (long)blockIdx.z * sBz;
  const long cz = (long)blockIdx.z * sCz;

  v8f a00 = {}, a01 = {}, a10 = {}, a11 = {};

  for (int k0 = 0; k0 < K; k0 += 32) {
    // ---------------- stage A (128 x 32) ----------------
    if (aMode == 0) {                      // contiguous along k
#pragma unroll
      for (int r = 0; r < 2; ++r) {
        const int cid = tid + 256 * r;     // 512 chunks of 8
        const int mm  = cid >> 2;          // 0..127
        const int kq  = (cid & 3) * 8;     // 0,8,16,24
        const float* ap = A + az + (long)(mBase + mm) * sAm + (k0 + kq);
        f32x4 v0 = *(const f32x4*)(ap);
        f32x4 v1 = *(const f32x4*)(ap + 4);
        *(u16x8*)&Alds[mm * 40 + kq] = pack8(v0, v1);
      }
    } else {                               // contiguous along m (Q^T)
      const int mq = (tid & 31) * 4;       // 0..124
      const int kk = tid >> 5;             // 0..7
#pragma unroll
      for (int r = 0; r < 4; ++r) {
        const int k = kk + r * 8;
        const float* ap = A + az + (long)(k0 + k) * sAk + (mBase + mq);
        f32x4 v = *(const f32x4*)ap;
        Alds[(mq + 0) * 40 + k] = f2bf(v.x);
        Alds[(mq + 1) * 40 + k] = f2bf(v.y);
        Alds[(mq + 2) * 40 + k] = f2bf(v.z);
        Alds[(mq + 3) * 40 + k] = f2bf(v.w);
      }
    }
    // ---------------- stage B transposed: Blds[n][k] (64 x 32) ----------------
    if (bMode == 1) {                      // contiguous along k
      const int nn = tid >> 2;             // 0..63
      const int kq = (tid & 3) * 8;
      const float* bp = B + bz + (long)(nBase + nn) * sBn + (k0 + kq);
      f32x4 v0 = *(const f32x4*)(bp);
      f32x4 v1 = *(const f32x4*)(bp + 4);
      *(u16x8*)&Blds[nn * 40 + kq] = pack8(v0, v1);
    } else if (bMode == 0) {               // contiguous along n (activations)
      const int nq = (tid & 15) * 4;
      const int kk = tid >> 4;
#pragma unroll
      for (int hh = 0; hh < 2; ++hh) {
        const int k = kk + hh * 16;
        const float* bp = B + bz + (long)(k0 + k) * sBk + (nBase + nq);
        f32x4 v = *(const f32x4*)bp;
        Blds[(nq + 0) * 40 + k] = f2bf(v.x);
        Blds[(nq + 1) * 40 + k] = f2bf(v.y);
        Blds[(nq + 2) * 40 + k] = f2bf(v.z);
        Blds[(nq + 3) * 40 + k] = f2bf(v.w);
      }
    } else {                               // conv3 im2col
      const int nq = (tid & 15) * 4;
      const int kk = tid >> 4;
#pragma unroll
      for (int hh = 0; hh < 2; ++hh) {
        const int kl = kk + hh * 16;
        const int k  = k0 + kl;
        const int c  = k / 3;
        const int j  = k - 3 * c;
        const float* bp = B + bz + (long)c * sBk;
#pragma unroll
        for (int i = 0; i < 4; ++i) {
          const int col = nBase + nq + i + j - 1;
          const float v = (col >= 0 && col < convN) ? bp[col] : 0.f;
          Blds[(nq + i) * 40 + kl] = f2bf(v);
        }
      }
    }
    __syncthreads();

    // ---------------- fragments + 4 WMMAs (2x2) ----------------
    FragBF fa0, fa1, fb0, fb1;
    const unsigned short* ap0 = &Alds[(mGrp * 32 + l15) * 40 + half * 8];
    fa0.h[0] = *(const u16x8*)(ap0);
    fa0.h[1] = *(const u16x8*)(ap0 + 16);
    const unsigned short* ap1 = ap0 + 16 * 40;
    fa1.h[0] = *(const u16x8*)(ap1);
    fa1.h[1] = *(const u16x8*)(ap1 + 16);
    const unsigned short* bp0 = &Blds[(nGrp * 32 + l15) * 40 + half * 8];
    fb0.h[0] = *(const u16x8*)(bp0);
    fb0.h[1] = *(const u16x8*)(bp0 + 16);
    const unsigned short* bp1 = bp0 + 16 * 40;
    fb1.h[0] = *(const u16x8*)(bp1);
    fb1.h[1] = *(const u16x8*)(bp1 + 16);

    a00 = __builtin_amdgcn_wmma_f32_16x16x32_bf16(false, fa0.v, false, fb0.v, (short)0, a00, false, false);
    a01 = __builtin_amdgcn_wmma_f32_16x16x32_bf16(false, fa0.v, false, fb1.v, (short)0, a01, false, false);
    a10 = __builtin_amdgcn_wmma_f32_16x16x32_bf16(false, fa1.v, false, fb0.v, (short)0, a10, false, false);
    a11 = __builtin_amdgcn_wmma_f32_16x16x32_bf16(false, fa1.v, false, fb1.v, (short)0, a11, false, false);

    __syncthreads();
  }

  // ---------------- epilogue ----------------
  const int gn0 = nBase + nGrp * 32 + l15;
  const int gn1 = gn0 + 16;
#pragma unroll
  for (int i = 0; i < 8; ++i) {
    const int gm0 = mBase + mGrp * 32 + half * 8 + i;
    const int gm1 = gm0 + 16;
    const float bv0 = bias ? bias[gm0] : 0.f;
    const float bv1 = bias ? bias[gm1] : 0.f;
    float v00 = alpha * a00[i] + bv0;
    float v01 = alpha * a01[i] + bv0;
    float v10 = alpha * a10[i] + bv1;
    float v11 = alpha * a11[i] + bv1;
    if (relu) {
      v00 = fmaxf(v00, 0.f); v01 = fmaxf(v01, 0.f);
      v10 = fmaxf(v10, 0.f); v11 = fmaxf(v11, 0.f);
    }
    Cmat[(long)gm0 * sCm + (long)gn0 * sCn + cz] = v00;
    Cmat[(long)gm0 * sCm + (long)gn1 * sCn + cz] = v01;
    Cmat[(long)gm1 * sCm + (long)gn0 * sCn + cz] = v10;
    Cmat[(long)gm1 * sCm + (long)gn1 * sCn + cz] = v11;
  }
}

// =====================================================================
// scores[z, t, t+p-WIN] += sum_d (alpha * q[z, d, t]) * ek[p, d]
// =====================================================================
__global__ void rel_k_bias(const float* __restrict__ q,
                           const float* __restrict__ ek,
                           float* __restrict__ scores, float alpha)
{
  __shared__ float qv[DK];
  const int t = blockIdx.x;
  const int z = blockIdx.y;            // b*H + h
  const int d = threadIdx.x;           // 64 threads
  qv[d] = alpha * q[(long)z * DK * T + (long)d * T + t];
  __syncthreads();
  if (d < 2 * WIN + 1) {
    int s = t + d - WIN;
    if (s >= 0 && s < T) {
      float a = 0.f;
#pragma unroll
      for (int dd = 0; dd < DK; ++dd) a += qv[dd] * ek[d * DK + dd];
      scores[(long)z * T * T + (long)t * T + s] += a;
    }
  }
}

// =====================================================================
// masked softmax over one score row (in place, fp32)
// =====================================================================
__global__ __launch_bounds__(256)
void softmax_row(float* __restrict__ scores, const float* __restrict__ xmask)
{
  __shared__ float red[256];
  const int t = blockIdx.x;
  const int z = blockIdx.y;
  const int b = z / H;
  const int tid = threadIdx.x;
  float* row = scores + (long)z * T * T + (long)t * T;
  const float mt = xmask[(long)b * T + t];

  float mx = -1e30f;
  for (int s = tid; s < T; s += 256) {
    float v = row[s];
    if (mt * xmask[(long)b * T + s] == 0.f) v = -10000.f;
    mx = fmaxf(mx, v);
  }
  red[tid] = mx; __syncthreads();
  for (int o = 128; o > 0; o >>= 1) { if (tid < o) red[tid] = fmaxf(red[tid], red[tid + o]); __syncthreads(); }
  mx = red[0]; __syncthreads();

  float sum = 0.f;
  for (int s = tid; s < T; s += 256) {
    float v = row[s];
    if (mt * xmask[(long)b * T + s] == 0.f) v = -10000.f;
    float e = __expf(v - mx);
    row[s] = e;
    sum += e;
  }
  red[tid] = sum; __syncthreads();
  for (int o = 128; o > 0; o >>= 1) { if (tid < o) red[tid] += red[tid + o]; __syncthreads(); }
  const float rs = 1.f / red[0];
  for (int s = tid; s < T; s += 256) row[s] *= rs;
}

// =====================================================================
// ctx[z, d, t] += sum_p P[z, t, t+p-WIN] * ev[p, d]
// =====================================================================
__global__ void rel_v_add(const float* __restrict__ P,
                          const float* __restrict__ ev,
                          float* __restrict__ ctx)
{
  const int t = blockIdx.x;
  const int z = blockIdx.y;
  const int d = threadIdx.x;           // 64 threads
  float a = 0.f;
#pragma unroll
  for (int p = 0; p < 2 * WIN + 1; ++p) {
    int s = t + p - WIN;
    if (s >= 0 && s < T)
      a += P[(long)z * T * T + (long)t * T + s] * ev[p * DK + d];
  }
  ctx[(long)z * DK * T + (long)d * T + t] += a;
}

// =====================================================================
// out[b, :, t] = LayerNorm_channel(x[b,:,t] + y[b,:,t]) * g + beta
// =====================================================================
__global__ __launch_bounds__(128)
void ln_add(const float* __restrict__ x, const float* __restrict__ y,
            const float* __restrict__ g, const float* __restrict__ beta,
            float* __restrict__ out)
{
  __shared__ float r1[128], r2[128];
  const int t = blockIdx.x;
  const int b = blockIdx.y;
  const int tid = threadIdx.x;
  const long base = (long)b * C * T + t;

  float s = 0.f, s2 = 0.f;
  for (int c = tid; c < C; c += 128) {
    float v = x[base + (long)c * T] + y[base + (long)c * T];
    s += v; s2 += v * v;
  }
  r1[tid] = s; r2[tid] = s2; __syncthreads();
  for (int o = 64; o > 0; o >>= 1) {
    if (tid < o) { r1[tid] += r1[tid + o]; r2[tid] += r2[tid + o]; }
    __syncthreads();
  }
  const float mean = r1[0] / (float)C;
  const float var  = r2[0] / (float)C - mean * mean;
  const float rstd = rsqrtf(var + EPS);
  for (int c = tid; c < C; c += 128) {
    float v = x[base + (long)c * T] + y[base + (long)c * T];
    out[base + (long)c * T] = (v - mean) * rstd * g[c] + beta[c];
  }
}

// =====================================================================
// coupling + logdet
// =====================================================================
__global__ void zero_logdet(float* ld) { if (threadIdx.x < BATCH) ld[threadIdx.x] = 0.f; }

__global__ __launch_bounds__(256)
void coupling(const float* __restrict__ xin, const float* __restrict__ stats,
              float* __restrict__ out, float* __restrict__ logdet)
{
  __shared__ float red[256];
  const int tid = threadIdx.x;
  const int idx = blockIdx.x * 256 + tid;      // over HALF*T
  const int b   = blockIdx.y;
  const int j   = idx / T;
  const int t   = idx - j * T;
  const long baseX = (long)b * 2 * HALF * T;

  float x0 = xin[baseX + (long)j * T + t];
  float x1 = xin[baseX + (long)(HALF + j) * T + t];
  float m  = stats[baseX + (long)j * T + t];
  float lg = stats[baseX + (long)(HALF + j) * T + t];
  out[baseX + (long)j * T + t]          = x0;
  out[baseX + (long)(HALF + j) * T + t] = m + x1 * __expf(lg);

  red[tid] = lg; __syncthreads();
  for (int o = 128; o > 0; o >>= 1) { if (tid < o) red[tid] += red[tid + o]; __syncthreads(); }
  if (tid == 0) atomicAdd(&logdet[b], red[0]);
}

// =====================================================================
// host orchestration
// =====================================================================
static inline void run_gemm(hipStream_t st,
    const float* A, long sAm, long sAk, long sAz,
    const float* B, long sBk, long sBn, long sBz,
    const float* bias,
    float* Cm, long sCm, long sCn, long sCz,
    int M, int N, int K, int Z,
    float alpha, int relu, int aMode, int bMode, int convN)
{
  dim3 grid(N / 64, M / 128, Z);
  gemm_bf16_wmma<<<grid, dim3(256), 0, st>>>(A, sAm, sAk, sAz, B, sBk, sBn, sBz,
                                             bias, Cm, sCm, sCn, sCz,
                                             M, N, K, alpha, relu, aMode, bMode, convN);
}

extern "C" void kernel_launch(void* const* d_in, const int* in_sizes, int n_in,
                              void* d_out, int out_size, void* d_ws, size_t ws_size,
                              hipStream_t stream)
{
  const float* x      = (const float*)d_in[0];
  const float* xmask  = (const float*)d_in[1];
  const float* pre_w  = (const float*)d_in[2];
  const float* pre_b  = (const float*)d_in[3];
  const float* post_w = (const float*)d_in[4];
  const float* post_b = (const float*)d_in[5];
  const float* qw = (const float*)d_in[6];  const float* qb = (const float*)d_in[7];
  const float* kw = (const float*)d_in[8];  const float* kb = (const float*)d_in[9];
  const float* vw = (const float*)d_in[10]; const float* vb = (const float*)d_in[11];
  const float* ow = (const float*)d_in[12]; const float* ob = (const float*)d_in[13];
  const float* ek = (const float*)d_in[14]; const float* ev = (const float*)d_in[15];
  const float* g1 = (const float*)d_in[16]; const float* b1 = (const float*)d_in[17];
  const float* g2 = (const float*)d_in[18]; const float* b2 = (const float*)d_in[19];
  const float* f1w = (const float*)d_in[20]; const float* f1b = (const float*)d_in[21];
  const float* f2w = (const float*)d_in[22]; const float* f2b = (const float*)d_in[23];

  // workspace carve-out (floats)
  const size_t NCT = (size_t)BATCH * C * T;
  float* w = (float*)d_ws;
  float* xf     = w;            w += NCT;
  float* qbuf   = w;            w += NCT;
  float* kbuf   = w;            w += NCT;
  float* vbuf   = w;            w += NCT;
  float* ctx    = w;            w += NCT;
  float* ybuf   = w;            w += NCT;
  float* stats  = w;            w += NCT;
  float* hbuf   = w;            w += (size_t)BATCH * FF * T;
  float* scores = w;            // BATCH*H*T*T

  const long CT   = (long)C * T;
  const long X2T  = (long)2 * HALF * T;  // input batch stride
  const float aQK = 0.125f;              // DK^-0.5
  const int BH = BATCH * H;

  // h = pre_w @ x0 + pre_b   (x_mask == 1)
  run_gemm(stream, pre_w, HALF, 1, 0,
           x, T, 1, X2T, pre_b,
           xf, T, 1, CT, C, T, HALF, BATCH, 1.f, 0, 0, 0, T);

  for (int i = 0; i < LAYERS; ++i) {
    const float* qwL = qw + (size_t)i * C * C;  const float* qbL = qb + (size_t)i * C;
    const float* kwL = kw + (size_t)i * C * C;  const float* kbL = kb + (size_t)i * C;
    const float* vwL = vw + (size_t)i * C * C;  const float* vbL = vb + (size_t)i * C;
    const float* owL = ow + (size_t)i * C * C;  const float* obL = ob + (size_t)i * C;
    const float* ekL = ek + (size_t)i * (2 * WIN + 1) * DK;
    const float* evL = ev + (size_t)i * (2 * WIN + 1) * DK;

    // Q/K/V projections
    run_gemm(stream, qwL, C, 1, 0, xf, T, 1, CT, qbL, qbuf, T, 1, CT, C, T, C, BATCH, 1.f, 0, 0, 0, T);
    run_gemm(stream, kwL, C, 1, 0, xf, T, 1, CT, kbL, kbuf, T, 1, CT, C, T, C, BATCH, 1.f, 0, 0, 0, T);
    run_gemm(stream, vwL, C, 1, 0, xf, T, 1, CT, vbL, vbuf, T, 1, CT, C, T, C, BATCH, 1.f, 0, 0, 0, T);

    // scores[z,t,s] = alpha * sum_d q[z,d,t] k[z,d,s]   (z = b*H+h)
    run_gemm(stream, qbuf, 1, T, (long)DK * T,
             kbuf, T, 1, (long)DK * T, nullptr,
             scores, T, 1, (long)T * T, T, T, DK, BH, aQK, 0, /*aMode=*/1, /*bMode=*/0, T);

    rel_k_bias<<<dim3(T, BH), 64, 0, stream>>>(qbuf, ekL, scores, aQK);
    softmax_row<<<dim3(T, BH), 256, 0, stream>>>(scores, xmask);

    // ctx^T: ctx[z,d,t] = sum_s P[z,t,s] v[z,d,s]
    //   A = P (m=t, k=s, k-contiguous), B = V^T (k=s, n=d, k-contiguous),
    //   C strides transposed (sCm=1 over t, sCn=T over d)
    run_gemm(stream, scores, T, 1, (long)T * T,
             vbuf, 1, T, (long)DK * T, nullptr,
             ctx, 1, T, (long)DK * T, T, DK, T, BH, 1.f, 0, /*aMode=*/0, /*bMode=*/1, T);

    rel_v_add<<<dim3(T, BH), 64, 0, stream>>>(scores, evL, ctx);

    // output projection (ctx contiguous == (B, C, T))
    run_gemm(stream, owL, C, 1, 0, ctx, T, 1, CT, obL, ybuf, T, 1, CT, C, T, C, BATCH, 1.f, 0, 0, 0, T);

    ln_add<<<dim3(T, BATCH), 128, 0, stream>>>(xf, ybuf, g1 + (size_t)i * C, b1 + (size_t)i * C, xf);

    // FFN conv1 (KS=3, relu)
    run_gemm(stream, f1w + (size_t)i * FF * C * KS, (long)C * KS, 1, 0,
             xf, T, 1, CT, f1b + (size_t)i * FF,
             hbuf, T, 1, (long)FF * T, FF, T, C * KS, BATCH, 1.f, 1, 0, /*bMode=*/2, T);
    // FFN conv2 (KS=3)
    run_gemm(stream, f2w + (size_t)i * C * FF * KS, (long)FF * KS, 1, 0,
             hbuf, T, 1, (long)FF * T, f2b + (size_t)i * C,
             ybuf, T, 1, CT, C, T, FF * KS, BATCH, 1.f, 0, 0, /*bMode=*/2, T);

    ln_add<<<dim3(T, BATCH), 128, 0, stream>>>(xf, ybuf, g2 + (size_t)i * C, b2 + (size_t)i * C, xf);
  }

  // stats = post_w @ h + post_b
  run_gemm(stream, post_w, C, 1, 0, xf, T, 1, CT, post_b,
           stats, T, 1, CT, 2 * HALF, T, C, BATCH, 1.f, 0, 0, 0, T);

  // coupling + logdet
  float* outp   = (float*)d_out;
  float* logdet = outp + (size_t)BATCH * 2 * HALF * T;
  zero_logdet<<<1, 32, 0, stream>>>(logdet);
  coupling<<<dim3((HALF * T) / 256, BATCH), 256, 0, stream>>>(x, stats, outp, logdet);
}
